// SparseWindowAttention_22591527977127
// MI455X (gfx1250) — matrix operand order repaired and checked by reading, not compiled, not run on hardware
//
#include <hip/hip_runtime.h>

// ---------------------------------------------------------------------------
// Problem constants (b = 1)
// ---------------------------------------------------------------------------
#define T_   6
#define H_   40
#define W_   72
#define D_   512
#define NH_  4
#define HD_  128
#define WH_  5
#define WW_  9
#define S_   45            // tokens per window per frame
#define NWX  8
#define NWIN 64
#define KPW  373           // keys per window per frame: 45 + 148 + 180
#define P_   180           // pooled tokens per frame (10*18)
#define NTOK   (T_ * H_ * W_)   // 17280
#define NPTOK  (T_ * P_)        // 1080

typedef __attribute__((ext_vector_type(16))) __bf16      v16bf;
typedef __attribute__((ext_vector_type(8)))  float       v8f;
typedef __attribute__((ext_vector_type(4)))  unsigned int u32x4;

union V16U { v16bf v; u32x4 q[2]; __bf16 e[16]; };

// D = A(16x32 bf16) * B(32x16 bf16) + C(16x16 f32)
__device__ __forceinline__ v8f wmma_bf16(v16bf a, v16bf b, v8f c) {
  return __builtin_amdgcn_wmma_f32_16x16x32_bf16(false, a, false, b, (short)0, c,
                                                 false, false);
}

// A-fragment, 16x32, row-major source: lane<16 -> M=lane, k = kb+0..15
//                                      lane>=16 -> M=lane-16, k = kb+16..31
__device__ __forceinline__ v16bf load_a16(const __bf16* row, int kb, int half) {
  V16U u;
  const int o = kb + (half ? 16 : 0);
  u.q[0] = *reinterpret_cast<const u32x4*>(row + o);
  u.q[1] = *reinterpret_cast<const u32x4*>(row + o + 8);
  return u.v;
}

// B-fragment, 32x16: lane%16 selects the N column (row ptr passed per lane),
// elements are contiguous K: lane<16 -> k = kb+0..15, lane>=16 -> k = kb+16..31
__device__ __forceinline__ v16bf load_b16(const __bf16* row, int kb, int half) {
  V16U u;
  const int o = kb + (half ? 16 : 0);
  u.q[0] = *reinterpret_cast<const u32x4*>(row + o);
  u.q[1] = *reinterpret_cast<const u32x4*>(row + o + 8);
  return u.v;
}

// Wave-private LDS producer/consumer: same-wave DS ops are in-order; just stop
// the compiler from reordering around the exchange.
__device__ __forceinline__ void wave_sync() {
  __builtin_amdgcn_fence(__ATOMIC_RELEASE, "wavefront");
  __builtin_amdgcn_wave_barrier();
  __builtin_amdgcn_fence(__ATOMIC_ACQUIRE, "wavefront");
}

// ---------------------------------------------------------------------------
// Packing / prep kernels
// ---------------------------------------------------------------------------
__global__ void cvt_bf16(const float* __restrict__ src, __bf16* __restrict__ dst, int n) {
  int i = blockIdx.x * blockDim.x + threadIdx.x;
  if (i < n) dst[i] = (__bf16)src[i];
}

// W[k][n] (512x512) -> Wt[n][k] bf16 so B-fragments are contiguous loads
__global__ void transpose512(const float* __restrict__ W, __bf16* __restrict__ Wt) {
  int i = blockIdx.x * blockDim.x + threadIdx.x;
  if (i >= 512 * 512) return;
  int nn = i >> 9, k = i & 511;
  Wt[(size_t)nn * 512 + k] = (__bf16)W[(size_t)k * 512 + nn];
}

// Depthwise 4x4/stride-4 pool conv + bias -> bf16 pooled tokens (t, 10, 18, 512)
__global__ void pool_conv(const float* __restrict__ x, const float* __restrict__ pw,
                          const float* __restrict__ pbias, __bf16* __restrict__ pxb) {
  int i = blockIdx.x * blockDim.x + threadIdx.x;
  if (i >= NPTOK * D_) return;
  int c = i & 511, r = i >> 9;
  int px = r % 18; r /= 18;
  int py = r % 10; int bt = r / 10;
  float s = pbias[c];
#pragma unroll
  for (int ky = 0; ky < 4; ++ky)
#pragma unroll
    for (int kx = 0; kx < 4; ++kx)
      s += x[((size_t)((bt * H_ + py * 4 + ky) * W_ + px * 4 + kx)) * D_ + c] *
           pw[(ky * 4 + kx) * D_ + c];
  pxb[i] = (__bf16)s;
}

// Window is "masked" iff any mask element inside it (any frame) > 0
__global__ void wmask_kernel(const float* __restrict__ masks, int* __restrict__ wmask) {
  int n = threadIdx.x;
  if (n >= NWIN) return;
  int wy = n / NWX, wx = n % NWX;
  int any = 0;
  for (int f = 0; f < T_; ++f)
    for (int i = 0; i < WH_; ++i)
      for (int j = 0; j < WW_; ++j)
        any |= (masks[(size_t)((f * H_ + wy * WH_ + i) * W_ + wx * WW_ + j)] > 0.f) ? 1 : 0;
  wmask[n] = any;
}

// Per-window key spatial index map: 45 window tokens, 148 rolled-valid tokens,
// 180 pooled tokens (encoded as 2880 + pool_index)
__global__ void idxmap_kernel(int* __restrict__ idxmap) {
  int n = threadIdx.x;
  if (n >= NWIN) return;
  int wy = n / NWX, wx = n % NWX;
  int* o = idxmap + n * KPW;
  int c = 0;
  for (int s = 0; s < S_; ++s) {
    int i = s / WW_, j = s % WW_;
    o[c++] = (wy * WH_ + i) * W_ + wx * WW_ + j;
  }
  const int dy[4] = {3, 3, -3, -3};
  const int dx[4] = {5, -5, 5, -5};
  for (int q = 0; q < 4; ++q)
    for (int s = 0; s < S_; ++s) {
      int i = s / WW_, j = s % WW_;
      bool inval = (q == 0 && i < 2 && j < 4) || (q == 1 && i < 2 && j >= 5) ||
                   (q == 2 && i >= 3 && j < 4) || (q == 3 && i >= 3 && j >= 5);
      if (!inval) {
        int y  = (wy * WH_ + i + dy[q] + H_) % H_;
        int xx = (wx * WW_ + j + dx[q] + W_) % W_;
        o[c++] = y * W_ + xx;
      }
    }
  for (int jp = 0; jp < P_; ++jp) o[c++] = H_ * W_ + jp;
}

// ---------------------------------------------------------------------------
// WMMA GEMM: C[M][512] = A[M][512] * Bt^T + bias.
// 4 waves; each wave owns a 32x64 output tile (two 16-row M tiles reusing the
// same B fragments). Fully unrolled, explicitly double-buffered K loop: loads
// for step k+1 are issued before the WMMAs of step k, and full unrolling turns
// the buffer swap into SSA renames (no v_mov churn between WMMAs).
// ---------------------------------------------------------------------------
__global__ __launch_bounds__(128) void gemm512(
    const __bf16* __restrict__ A, const __bf16* __restrict__ Bt,
    const float* __restrict__ bias, __bf16* __restrict__ outB,
    float* __restrict__ outF, int M) {
  const int wave = threadIdx.x >> 5;
  const int lane = threadIdx.x & 31;
  const int half = lane >> 4, l16 = lane & 15;
  const int m0 = blockIdx.x * 128 + wave * 32;
  const int n0 = blockIdx.y * 64;
  int mr0 = m0 + l16;
  int mr1 = m0 + 16 + l16;
  if (mr0 >= M) mr0 = M - 1;
  if (mr1 >= M) mr1 = M - 1;
  const __bf16* arow0 = A + (size_t)mr0 * 512;
  const __bf16* arow1 = A + (size_t)mr1 * 512;
  const __bf16* brow[4];
#pragma unroll
  for (int nt = 0; nt < 4; ++nt) brow[nt] = Bt + (size_t)(n0 + nt * 16 + l16) * 512;

  v8f acc[2][4];
#pragma unroll
  for (int mt = 0; mt < 2; ++mt)
#pragma unroll
    for (int nt = 0; nt < 4; ++nt)
#pragma unroll
      for (int r = 0; r < 8; ++r) acc[mt][nt][r] = 0.f;

  // prologue: fragments for k-step 0
  v16bf a0 = load_a16(arow0, 0, half);
  v16bf a1 = load_a16(arow1, 0, half);
  v16bf bf[4];
#pragma unroll
  for (int nt = 0; nt < 4; ++nt) bf[nt] = load_b16(brow[nt], 0, half);

#pragma unroll
  for (int kbs = 0; kbs < 512; kbs += 32) {
    v16bf a0n, a1n, bn[4];
    const bool more = (kbs + 32 < 512);  // compile-time per unrolled iteration
    if (more) {
      // issue next k-step loads before consuming current fragments
      a0n = load_a16(arow0, kbs + 32, half);
      a1n = load_a16(arow1, kbs + 32, half);
#pragma unroll
      for (int nt = 0; nt < 4; ++nt) bn[nt] = load_b16(brow[nt], kbs + 32, half);
    }
#pragma unroll
    for (int nt = 0; nt < 4; ++nt) {
      acc[0][nt] = wmma_bf16(a0, bf[nt], acc[0][nt]);
      acc[1][nt] = wmma_bf16(a1, bf[nt], acc[1][nt]);
    }
    if (more) {
      a0 = a0n;
      a1 = a1n;
#pragma unroll
      for (int nt = 0; nt < 4; ++nt) bf[nt] = bn[nt];
    }
  }

#pragma unroll
  for (int mt = 0; mt < 2; ++mt)
#pragma unroll
    for (int nt = 0; nt < 4; ++nt)
#pragma unroll
      for (int r = 0; r < 8; ++r) {
        const int m = m0 + mt * 16 + r + half * 8;  // C layout: VGPR r -> row
        if (m < M) {
          const int nn = n0 + nt * 16 + l16;
          const float v = acc[mt][nt][r] + bias[nn];
          if (outB) outB[(size_t)m * 512 + nn] = (__bf16)v;
          else      outF[(size_t)m * 512 + nn] = v;
        }
      }
}

// ---------------------------------------------------------------------------
// Flash attention over one 16-query tile of one (window, head).
// frame < 0: dense mode (queries/keys span all frames, keys per frame = KPW)
// frame >= 0: local per-frame mode (45x45, keys are the window tokens)
// ---------------------------------------------------------------------------
__device__ void flash_tile(const __bf16* __restrict__ qb,
                           const __bf16* __restrict__ kb,
                           const __bf16* __restrict__ vb,
                           const int* __restrict__ idxw,
                           __bf16* __restrict__ attnb,
                           __bf16* pb, int* ob,
                           int head, int wy, int wx,
                           int q0, int nq, int nk, int frame) {
  const int lane = threadIdx.x & 31;
  const int half = lane >> 4;
  const int l16  = lane & 15;
  const float scale = 0.088388347648318447f;  // 1/sqrt(128)

  // --- Q A-fragments (held in registers for the whole key sweep) ---
  int m = q0 + l16;
  if (m >= nq) m = nq - 1;  // clamp pad rows (results never stored)
  const int qf = (frame < 0) ? (m / S_) : frame;
  const int qs = (frame < 0) ? (m % S_) : m;
  const int qy = wy * WH_ + qs / WW_;
  const int qx = wx * WW_ + qs % WW_;
  const __bf16* qrow = qb + ((size_t)((qf * H_ + qy) * W_ + qx)) * D_ + head * HD_;
  v16bf qa[4];
#pragma unroll
  for (int d = 0; d < 4; ++d) qa[d] = load_a16(qrow, d * 32, half);

  v8f acc[8];                 // O accumulator: 8 N-tiles of 16 dims
  float rmax[8], rsum[8];
#pragma unroll
  for (int nt = 0; nt < 8; ++nt)
#pragma unroll
    for (int r = 0; r < 8; ++r) acc[nt][r] = 0.f;
#pragma unroll
  for (int r = 0; r < 8; ++r) { rmax[r] = -1e30f; rsum[r] = 0.f; }

  // key-row element offset for this lane's key of a chunk
  auto key_off = [&](int ki) -> int {
    int koff = head * HD_;
    if (ki < nk) {
      const int kf = (frame < 0) ? (ki / KPW) : frame;
      const int kj = (frame < 0) ? (ki % KPW) : ki;
      const int p  = idxw[kj];
      koff += (p < H_ * W_) ? (kf * (H_ * W_) + p) * D_
                            : NTOK * D_ + (kf * P_ + (p - H_ * W_)) * D_;
    }
    return koff;
  };
  int koff = key_off(lane);

  for (int cbs = 0; cbs < nk; cbs += 32) {
    ob[lane] = koff;
    wave_sync();

    // pipeline: compute next chunk's offset now and prefetch its K/V rows
    // (global_prefetch_b8) while this chunk's WMMAs run
    if (cbs + 32 < nk) {
      koff = key_off(cbs + 32 + lane);
      __builtin_prefetch(kb + koff, 0, 3);
      __builtin_prefetch(vb + koff, 0, 3);
    }

    // --- S = Q * K^T for two 16-key tiles ---
    v8f st[2];
#pragma unroll
    for (int tt = 0; tt < 2; ++tt) {
      const __bf16* krow = kb + ob[tt * 16 + l16];
      v8f c;
#pragma unroll
      for (int r = 0; r < 8; ++r) c[r] = 0.f;
#pragma unroll
      for (int d = 0; d < 4; ++d)
        c = wmma_bf16(qa[d], load_b16(krow, d * 32, half), c);
      const int ci = cbs + tt * 16 + l16;
      const float msk = (ci < nk) ? 0.f : -1e30f;
#pragma unroll
      for (int r = 0; r < 8; ++r) c[r] = c[r] * scale + msk;
      st[tt] = c;
    }

    // --- online softmax: row = VGPR index (+8 for upper half), cols across 16 lanes ---
#pragma unroll
    for (int r = 0; r < 8; ++r) {
      float mx = fmaxf(st[0][r], st[1][r]);
      mx = fmaxf(mx, __shfl_xor(mx, 1));
      mx = fmaxf(mx, __shfl_xor(mx, 2));
      mx = fmaxf(mx, __shfl_xor(mx, 4));
      mx = fmaxf(mx, __shfl_xor(mx, 8));
      const float nm = fmaxf(rmax[r], mx);
      const float al = __expf(rmax[r] - nm);
      rmax[r] = nm;
      const float p0 = __expf(st[0][r] - nm);
      const float p1 = __expf(st[1][r] - nm);
      st[0][r] = p0;
      st[1][r] = p1;
      float ls = p0 + p1;
      ls += __shfl_xor(ls, 1);
      ls += __shfl_xor(ls, 2);
      ls += __shfl_xor(ls, 4);
      ls += __shfl_xor(ls, 8);
      rsum[r] = rsum[r] * al + ls;
#pragma unroll
      for (int nt = 0; nt < 8; ++nt) acc[nt][r] *= al;
    }

    // --- C-layout P -> A-layout via wave-private LDS (16 x 32 bf16) ---
#pragma unroll
    for (int tt = 0; tt < 2; ++tt)
#pragma unroll
      for (int r = 0; r < 8; ++r)
        pb[(r + half * 8) * 32 + tt * 16 + l16] = (__bf16)st[tt][r];
    wave_sync();

    const v16bf pa = load_a16(pb + l16 * 32, 0, half);
    int voff[16];
#pragma unroll
    for (int e = 0; e < 16; ++e) voff[e] = ob[half * 16 + e];

    // --- O += P * V : B element e comes from key (half*16 + e), fixed dim column ---
#pragma unroll
    for (int nt = 0; nt < 8; ++nt) {
      V16U u;
      const int dcol = nt * 16 + l16;
#pragma unroll
      for (int e = 0; e < 16; ++e) u.e[e] = vb[(size_t)voff[e] + dcol];
      acc[nt] = wmma_bf16(pa, u.v, acc[nt]);
    }
  }

  // --- normalize + scatter to (t,h,w,c) bf16 attention output ---
#pragma unroll
  for (int r = 0; r < 8; ++r) {
    const int mm = q0 + r + half * 8;
    if (mm < nq) {
      const float inv = 1.f / rsum[r];
      const int of = (frame < 0) ? (mm / S_) : frame;
      const int os = (frame < 0) ? (mm % S_) : mm;
      const int oy = wy * WH_ + os / WW_;
      const int ox = wx * WW_ + os % WW_;
      __bf16* orow = attnb + ((size_t)((of * H_ + oy) * W_ + ox)) * D_ + head * HD_;
#pragma unroll
      for (int nt = 0; nt < 8; ++nt)
        orow[nt * 16 + l16] = (__bf16)(acc[nt][r] * inv);
    }
  }
}

__global__ __launch_bounds__(128) void attn_kernel(
    const __bf16* __restrict__ qb, const __bf16* __restrict__ kb,
    const __bf16* __restrict__ vb, const int* __restrict__ idxmap,
    const int* __restrict__ wmask, __bf16* __restrict__ attnb) {
  __shared__ __align__(16) __bf16 pbuf[4][16 * 32];
  __shared__ int offbuf[4][32];
  const int n    = blockIdx.x >> 2;
  const int head = blockIdx.x & 3;
  const int wy = n / NWX, wx = n % NWX;
  const int wave = threadIdx.x >> 5;
  const int* idxw = idxmap + n * KPW;
  __bf16* pb = pbuf[wave];
  int* ob = offbuf[wave];
  if (wmask[n]) {
    // dense: 270 queries -> 17 tiles of 16, keys = t*KPW = 2238
    for (int tile = wave; tile < 17; tile += 4)
      flash_tile(qb, kb, vb, idxw, attnb, pb, ob, head, wy, wx,
                 tile * 16, T_ * S_, T_ * KPW, -1);
  } else {
    // local: per frame, 45 queries (3 tiles) x 45 window keys
    for (int unit = wave; unit < 18; unit += 4)
      flash_tile(qb, kb, vb, idxw, attnb, pb, ob, head, wy, wx,
                 (unit % 3) * 16, S_, S_, unit / 3);
  }
}

// ---------------------------------------------------------------------------
// Host launch
// ---------------------------------------------------------------------------
extern "C" void kernel_launch(void* const* d_in, const int* in_sizes, int n_in,
                              void* d_out, int out_size, void* d_ws, size_t ws_size,
                              hipStream_t stream) {
  const float* x      = (const float*)d_in[0];
  const float* masks  = (const float*)d_in[1];
  const float* Wq     = (const float*)d_in[2];
  const float* bq     = (const float*)d_in[3];
  const float* Wk     = (const float*)d_in[4];
  const float* bk     = (const float*)d_in[5];
  const float* Wv     = (const float*)d_in[6];
  const float* bv     = (const float*)d_in[7];
  const float* Wp     = (const float*)d_in[8];
  const float* bp     = (const float*)d_in[9];
  const float* pool_w = (const float*)d_in[10];
  const float* pool_b = (const float*)d_in[11];

  char* ws = (char*)d_ws;
  auto alloc = [&](size_t bytes) -> void* {
    void* p = (void*)ws;
    ws += (bytes + 255) & ~(size_t)255;
    return p;
  };
  const size_t TOKB = (size_t)NTOK * D_ * sizeof(__bf16);
  const size_t KVB  = (size_t)(NTOK + NPTOK) * D_ * sizeof(__bf16);
  __bf16* xb    = (__bf16*)alloc(TOKB);
  __bf16* qb    = (__bf16*)alloc(TOKB);
  __bf16* kb    = (__bf16*)alloc(KVB);   // window k rows, then pooled k rows
  __bf16* vb    = (__bf16*)alloc(KVB);   // window v rows, then pooled v rows
  __bf16* attnb = (__bf16*)alloc(TOKB);
  __bf16* pxb   = (__bf16*)alloc((size_t)NPTOK * D_ * sizeof(__bf16));
  __bf16* Wqt   = (__bf16*)alloc((size_t)512 * 512 * sizeof(__bf16));
  __bf16* Wkt   = (__bf16*)alloc((size_t)512 * 512 * sizeof(__bf16));
  __bf16* Wvt   = (__bf16*)alloc((size_t)512 * 512 * sizeof(__bf16));
  __bf16* Wpt   = (__bf16*)alloc((size_t)512 * 512 * sizeof(__bf16));
  int* idxmap   = (int*)alloc((size_t)NWIN * KPW * sizeof(int));
  int* wmaskd   = (int*)alloc((size_t)NWIN * sizeof(int));
  (void)in_sizes; (void)n_in; (void)out_size; (void)ws_size;

  const int nx = NTOK * D_;
  cvt_bf16<<<(nx + 255) / 256, 256, 0, stream>>>(x, xb, nx);
  transpose512<<<1024, 256, 0, stream>>>(Wq, Wqt);
  transpose512<<<1024, 256, 0, stream>>>(Wk, Wkt);
  transpose512<<<1024, 256, 0, stream>>>(Wv, Wvt);
  transpose512<<<1024, 256, 0, stream>>>(Wp, Wpt);
  pool_conv<<<(NPTOK * D_ + 255) / 256, 256, 0, stream>>>(x, pool_w, pool_b, pxb);
  wmask_kernel<<<1, 64, 0, stream>>>(masks, wmaskd);
  idxmap_kernel<<<1, 64, 0, stream>>>(idxmap);

  dim3 g1((NTOK + 127) / 128, 8);
  gemm512<<<g1, 128, 0, stream>>>(xb, Wqt, bq, qb, nullptr, NTOK);
  gemm512<<<g1, 128, 0, stream>>>(xb, Wkt, bk, kb, nullptr, NTOK);
  gemm512<<<g1, 128, 0, stream>>>(xb, Wvt, bv, vb, nullptr, NTOK);
  dim3 g2((NPTOK + 127) / 128, 8);
  gemm512<<<g2, 128, 0, stream>>>(pxb, Wkt, bk, kb + (size_t)NTOK * D_, nullptr, NPTOK);
  gemm512<<<g2, 128, 0, stream>>>(pxb, Wvt, bv, vb + (size_t)NTOK * D_, nullptr, NPTOK);

  attn_kernel<<<NWIN * NH_, 128, 0, stream>>>(qb, kb, vb, idxmap, wmaskd, attnb);

  gemm512<<<g1, 128, 0, stream>>>(attnb, Wpt, bp, nullptr, (float*)d_out, NTOK);
}
